// SelfAttention_39204461478412
// MI455X (gfx1250) — matrix operand order repaired and checked by reading, not compiled, not run on hardware
//
#include <hip/hip_runtime.h>
#include <hip/hip_bf16.h>

typedef __attribute__((ext_vector_type(16))) __bf16 v16bf;
typedef __attribute__((ext_vector_type(8)))  __bf16 v8bf;
typedef __attribute__((ext_vector_type(8)))  float  v8f;

#define WMMA_BF16(a, b, c) \
  __builtin_amdgcn_wmma_f32_16x16x32_bf16(false, (a), false, (b), (short)0, (c), false, false)

// Problem constants: B=8, N=1024, C=768, H=12, hd=64; M = 8192 token rows.

// ---------------------------------------------------------------------------
// Async global -> LDS copy of 16B per lane (ASYNCcnt-tracked, no VGPR data).
// ---------------------------------------------------------------------------
__device__ __forceinline__ void async_b128(const __bf16* gsrc, __bf16* ldst) {
  const unsigned loff = (unsigned)(size_t)ldst;  // low 32 bits = LDS byte offset
  asm volatile("global_load_async_to_lds_b128 %0, %1, off"
               :: "v"(loff), "v"(gsrc) : "memory");
}

// ---------------------------------------------------------------------------
// Kernel 1: f32 -> bf16 convert (x and both weight matrices)
// ---------------------------------------------------------------------------
__global__ void cvt_f32_bf16(const float* __restrict__ in, __bf16* __restrict__ out, int n) {
  int i = blockIdx.x * blockDim.x + threadIdx.x;
  if (i < n) out[i] = (__bf16)in[i];
}

// ---------------------------------------------------------------------------
// Shared GEMM mainloop: C[256x64] = A[256xK] (bf16) @ W[64 rows x K] (bf16)^T
// 256 threads = 8 waves; wave w owns a 32-row strip (2 A frags x 4 B frags =
// 8 WMMAs per 32-K chunk). Double-buffered LDS fed by async LDS loads,
// manually unrolled 2x over the buffers.
// ---------------------------------------------------------------------------
__device__ __forceinline__ void gemm256x64_mainloop(
    const __bf16* __restrict__ A,   // [8192 x 768] bf16 row-major
    const __bf16* __restrict__ W,   // [Nall x 768] bf16 row-major
    int r0, int n0, v8f acc[8]) {
  __shared__ __attribute__((aligned(64))) __bf16 SA[2][256 * 32];
  __shared__ __attribute__((aligned(64))) __bf16 SB[2][64 * 32];

  const int tid  = threadIdx.x;
  const int lane = tid & 31;
  const int wave = tid >> 5;

  const v8f vzero = {};
#pragma unroll
  for (int g = 0; g < 8; ++g) acc[g] = vzero;

  const int brow = tid >> 2, bq = tid & 3;  // B staging: 4 threads/row, 16B each

  auto issue_stage = [&](int buf, int kc) {
    // A: one row per thread, 64B = 4 x b128
    const __bf16* ga = A + (size_t)(r0 + tid) * 768 + kc;
    __bf16* la = &SA[buf][tid * 32];
    async_b128(ga, la);
    async_b128(ga + 8, la + 8);
    async_b128(ga + 16, la + 16);
    async_b128(ga + 24, la + 24);
    // B: 64 rows x 32 k
    async_b128(W + (size_t)(n0 + brow) * 768 + kc + bq * 8,
               &SB[buf][brow * 32 + bq * 8]);
  };

  auto compute_chunk = [&](const __bf16* sa, const __bf16* sb) {
    const int klo   = (lane >> 4) * 16;
    const int mbase = wave * 32 + (lane & 15);
    v16bf a0, a1;
    {
      const v8bf x0 = *(const v8bf*)(sa + mbase * 32 + klo);
      const v8bf x1 = *(const v8bf*)(sa + mbase * 32 + klo + 8);
      const v8bf y0 = *(const v8bf*)(sa + (mbase + 16) * 32 + klo);
      const v8bf y1 = *(const v8bf*)(sa + (mbase + 16) * 32 + klo + 8);
#pragma unroll
      for (int j = 0; j < 8; ++j) {
        a0[j] = x0[j]; a0[j + 8] = x1[j];
        a1[j] = y0[j]; a1[j + 8] = y1[j];
      }
    }
    const int koff = (lane >> 4) * 16;
#pragma unroll
    for (int g = 0; g < 4; ++g) {
      const v16bf b = *(const v16bf*)(sb + (g * 16 + (lane & 15)) * 32 + koff);
      acc[g]     = WMMA_BF16(a0, b, acc[g]);
      acc[g + 4] = WMMA_BF16(a1, b, acc[g + 4]);
    }
  };

  issue_stage(0, 0);
  for (int p = 0; p < 12; ++p) {       // 24 K-chunks, 2 per trip
    const int kc = p * 64;
    // ---- buffer 0 ----
    issue_stage(1, kc + 32);
    asm volatile("s_wait_asynccnt 0x5" ::: "memory");  // buf0 landed, buf1 in flight
    __syncthreads();
    compute_chunk(SA[0], SB[0]);
    __syncthreads();
    // ---- buffer 1 ----
    if (p + 1 < 12) {
      issue_stage(0, kc + 64);
      asm volatile("s_wait_asynccnt 0x5" ::: "memory");
    } else {
      asm volatile("s_wait_asynccnt 0x0" ::: "memory");
    }
    __syncthreads();
    compute_chunk(SA[1], SB[1]);
    __syncthreads();
  }
}

// ---------------------------------------------------------------------------
// Kernel 2: QKV projection + fused RoPE + head-major scatter (+ q scaling)
// grid = (8192/256, 2304/64). Each 64-col block is one (section, head) pair.
// ---------------------------------------------------------------------------
__global__ void __launch_bounds__(256)
qkv_rope_kernel(const __bf16* __restrict__ xb, const __bf16* __restrict__ Wqkv,
                const float* __restrict__ sinp, const float* __restrict__ cosp,
                __bf16* __restrict__ qb, __bf16* __restrict__ kb, __bf16* __restrict__ vb) {
  v8f acc[8];
  const int r0 = blockIdx.x * 256;
  const int n0 = blockIdx.y * 64;
  gemm256x64_mainloop(xb, Wqkv, r0, n0, acc);

  const int lane = threadIdx.x & 31, wave = threadIdx.x >> 5;
  const int s  = n0 / 768;        // 0=q, 1=k, 2=v
  const int c0 = n0 % 768;
  const int h  = c0 / 64;
  __bf16* dst = (s == 0) ? qb : ((s == 1) ? kb : vb);
  const float qscale = (s == 0) ? 0.125f : 1.0f;  // hd^-0.5 folded into q

#pragma unroll
  for (int st = 0; st < 2; ++st) {
#pragma unroll
    for (int g = 0; g < 4; ++g) {
      const int j = g * 16 + (lane & 15);  // head-dim index 0..63
#pragma unroll
      for (int i = 0; i < 8; ++i) {
        const int r    = r0 + wave * 32 + st * 16 + i + ((lane >> 4) << 3);
        const int b    = r >> 10;
        const int nseq = r & 1023;
        float val = acc[st * 4 + g][i];
        if (s < 2) {  // RoPE: partner dim j^32 is in same lane, frag g^2
          const float cw = cosp[nseq * 64 + j];
          const float sw = sinp[nseq * 64 + j];
          const float partner = acc[st * 4 + (g ^ 2)][i];
          val = val * cw + ((g < 2) ? -partner : partner) * sw;
        }
        val *= qscale;
        dst[(size_t)((b * 12 + h) * 1024 + nseq) * 64 + j] = (__bf16)val;
      }
    }
  }
}

// ---------------------------------------------------------------------------
// Kernel 3: flash attention. grid = (1024/128, B*H). Wave owns 16 q rows.
// K tiles double-buffered via async LDS loads; V transposed via VALU staging.
// ---------------------------------------------------------------------------
__global__ void __launch_bounds__(256)
attn_kernel(const __bf16* __restrict__ qb, const __bf16* __restrict__ kb,
            const __bf16* __restrict__ vb, __bf16* __restrict__ attn) {
  __shared__ __attribute__((aligned(64))) __bf16 SK[2][32 * 64];   // keys x hd
  __shared__ __attribute__((aligned(64))) __bf16 SVt[2][64 * 32];  // hd x keys
  __shared__ __attribute__((aligned(64))) __bf16 SP[8][16 * 32];   // per-wave P

  const int tid = threadIdx.x, lane = tid & 31, wave = tid >> 5;
  const int bh = blockIdx.y;
  const int b = bh / 12, h = bh % 12;
  const size_t base = (size_t)bh * 1024 * 64;
  const int qr0 = blockIdx.x * 128 + wave * 16;

  auto issue_k = [&](int buf, int kt) {  // 32x64 bf16 tile is contiguous
    async_b128(kb + base + (size_t)kt * 64 + tid * 8, &SK[buf][tid * 8]);
  };
  auto stage_v = [&](int buf, int kt) {  // transpose: SVt[hd][key]
    const int kk = tid >> 3, cg = tid & 7;
    const v8bf v = *(const v8bf*)(vb + base + (size_t)(kt + kk) * 64 + cg * 8);
#pragma unroll
    for (int u = 0; u < 8; ++u) SVt[buf][(cg * 8 + u) * 32 + kk] = v[u];
  };

  // Load Q fragments for this wave's 16 rows (hd = two K-chunks of 32)
  v16bf qf[2];
  {
    const int m = lane & 15, klo = (lane >> 4) * 16;
#pragma unroll
    for (int cc = 0; cc < 2; ++cc)
#pragma unroll
      for (int j = 0; j < 16; ++j) {
        const int c = cc * 32 + klo + j;
        qf[cc][j] = qb[base + (size_t)(qr0 + m) * 64 + c];
      }
  }

  const v8f vzero = {};
  v8f O[4];
  float mrow[8], lrow[8];
#pragma unroll
  for (int g = 0; g < 4; ++g) O[g] = vzero;
#pragma unroll
  for (int i = 0; i < 8; ++i) { mrow[i] = -3.0e38f; lrow[i] = 0.f; }

  issue_k(0, 0);
  stage_v(0, 0);
  for (int it = 0; it < 32; ++it) {   // 1024 keys / 32
    const int cur = it & 1;
    const int kt  = it * 32;
    if (it + 1 < 32) {
      issue_k(cur ^ 1, kt + 32);
      asm volatile("s_wait_asynccnt 0x1" ::: "memory");
    } else {
      asm volatile("s_wait_asynccnt 0x0" ::: "memory");
    }
    __syncthreads();                  // cur SK/SVt visible to all waves
    if (it + 1 < 32) stage_v(cur ^ 1, kt + 32);  // overlaps with compute below

    // S = Q @ K^T : two 16x16 tiles covering 32 keys (q pre-scaled)
    v8f s0 = vzero, s1 = vzero;
    {
      const int kn = lane & 15, koff = (lane >> 4) * 16;
#pragma unroll
      for (int cc = 0; cc < 2; ++cc) {
        const v16bf b0 = *(const v16bf*)(&SK[cur][(0 * 16 + kn) * 64 + cc * 32 + koff]);
        const v16bf b1 = *(const v16bf*)(&SK[cur][(1 * 16 + kn) * 64 + cc * 32 + koff]);
        s0 = WMMA_BF16(qf[cc], b0, s0);
        s1 = WMMA_BF16(qf[cc], b1, s1);
      }
    }

    // Online softmax (D-tile rows live in 16-lane halves: xor masks 1..8)
#pragma unroll
    for (int i = 0; i < 8; ++i) {
      float mx = fmaxf(s0[i], s1[i]);
#pragma unroll
      for (int msk = 1; msk < 16; msk <<= 1) mx = fmaxf(mx, __shfl_xor(mx, msk, 32));
      const float mn    = fmaxf(mrow[i], mx);
      const float alpha = __expf(mrow[i] - mn);
      const float p0 = __expf(s0[i] - mn);
      const float p1 = __expf(s1[i] - mn);
      float rs = p0 + p1;
#pragma unroll
      for (int msk = 1; msk < 16; msk <<= 1) rs += __shfl_xor(rs, msk, 32);
      lrow[i] = lrow[i] * alpha + rs;
      mrow[i] = mn;
#pragma unroll
      for (int g = 0; g < 4; ++g) O[g][i] *= alpha;
      // transpose P (D layout -> A layout) via per-wave LDS scratch
      const int m = i + ((lane >> 4) << 3);
      SP[wave][m * 32 + (lane & 15)]      = (__bf16)p0;
      SP[wave][m * 32 + 16 + (lane & 15)] = (__bf16)p1;
    }
    asm volatile("s_wait_dscnt 0x0" ::: "memory");  // LDS RAW within wave

    // O += P @ V
    {
      const int m = lane & 15, klo = (lane >> 4) * 16;
      v16bf pf;
      const v8bf pA = *(const v8bf*)(&SP[wave][m * 32 + klo]);
      const v8bf pB = *(const v8bf*)(&SP[wave][m * 32 + klo + 8]);
#pragma unroll
      for (int j = 0; j < 8; ++j) { pf[j] = pA[j]; pf[j + 8] = pB[j]; }
      const int koff = (lane >> 4) * 16;
#pragma unroll
      for (int g = 0; g < 4; ++g) {
        const v16bf vf = *(const v16bf*)(&SVt[cur][(g * 16 + (lane & 15)) * 32 + koff]);
        O[g] = WMMA_BF16(pf, vf, O[g]);
      }
    }
    __syncthreads();  // all waves done with cur before it+2 overwrites it
  }

  // Normalize + scatter back to [B, N, H*hd]
#pragma unroll
  for (int g = 0; g < 4; ++g) {
    const int c = g * 16 + (lane & 15);
#pragma unroll
    for (int i = 0; i < 8; ++i) {
      const int m = i + ((lane >> 4) << 3);
      const int seq = qr0 + m;
      const float val = O[g][i] / lrow[i];
      attn[((size_t)(b * 1024 + seq)) * 768 + h * 64 + c] = (__bf16)val;
    }
  }
}

// ---------------------------------------------------------------------------
// Kernel 4: output projection + bias (f32 out). grid = (8192/256, 768/64).
// ---------------------------------------------------------------------------
__global__ void __launch_bounds__(256)
proj_kernel(const __bf16* __restrict__ attn, const __bf16* __restrict__ Wp,
            const float* __restrict__ bias, float* __restrict__ out) {
  v8f acc[8];
  const int r0 = blockIdx.x * 256;
  const int n0 = blockIdx.y * 64;
  gemm256x64_mainloop(attn, Wp, r0, n0, acc);

  const int lane = threadIdx.x & 31, wave = threadIdx.x >> 5;
#pragma unroll
  for (int st = 0; st < 2; ++st) {
#pragma unroll
    for (int g = 0; g < 4; ++g) {
      const int n = n0 + g * 16 + (lane & 15);
      const float bn = bias[n];
#pragma unroll
      for (int i = 0; i < 8; ++i) {
        const int r = r0 + wave * 32 + st * 16 + i + ((lane >> 4) << 3);
        out[(size_t)r * 768 + n] = acc[st * 4 + g][i] + bn;
      }
    }
  }
}

// ---------------------------------------------------------------------------
extern "C" void kernel_launch(void* const* d_in, const int* in_sizes, int n_in,
                              void* d_out, int out_size, void* d_ws, size_t ws_size,
                              hipStream_t stream) {
  (void)in_sizes; (void)n_in; (void)out_size; (void)ws_size;
  const float* x     = (const float*)d_in[0];
  const float* sinp  = (const float*)d_in[1];
  const float* cosp  = (const float*)d_in[2];
  const float* Wqkv  = (const float*)d_in[3];
  const float* Wproj = (const float*)d_in[4];
  const float* bproj = (const float*)d_in[5];
  float* out = (float*)d_out;

  const size_t SZ = (size_t)8192 * 768;      // elements per token-sized bf16 buffer
  __bf16* xb     = (__bf16*)d_ws;            // x in bf16
  __bf16* qb     = xb + SZ;                  // q [B,H,N,hd] (RoPE'd, pre-scaled)
  __bf16* kb     = qb + SZ;                  // k [B,H,N,hd] (RoPE'd)
  __bf16* vb     = kb + SZ;                  // v [B,H,N,hd]
  __bf16* ab     = vb + SZ;                  // attention out [B,N,C]
  __bf16* wqkvb  = ab + SZ;                  // W_qkv bf16 [2304 x 768]
  __bf16* wprojb = wqkvb + (size_t)2304 * 768;  // W_proj bf16 [768 x 768]

  const int nx = (int)SZ, nq = 2304 * 768, np = 768 * 768;
  cvt_f32_bf16<<<dim3((nx + 255) / 256), dim3(256), 0, stream>>>(x, xb, nx);
  cvt_f32_bf16<<<dim3((nq + 255) / 256), dim3(256), 0, stream>>>(Wqkv, wqkvb, nq);
  cvt_f32_bf16<<<dim3((np + 255) / 256), dim3(256), 0, stream>>>(Wproj, wprojb, np);

  qkv_rope_kernel<<<dim3(32, 36), dim3(256), 0, stream>>>(xb, wqkvb, sinp, cosp, qb, kb, vb);
  attn_kernel<<<dim3(8, 96), dim3(256), 0, stream>>>(qb, kb, vb, ab);
  proj_kernel<<<dim3(32, 12), dim3(256), 0, stream>>>(ab, wprojb, bproj, out);
}
